// HopfieldLayer_27805618275008
// MI455X (gfx1250) — matrix operand order repaired and checked by reading, not compiled, not run on hardware
//
#include <hip/hip_runtime.h>
#include <hip/hip_bf16.h>

#define N_FEATURES 64
#define N_CLASSES  1024
#define N_NEURONS  128
#define N_ITER     3
#define BATCH      256

typedef __attribute__((ext_vector_type(16))) _Float16 v16h;
typedef __attribute__((ext_vector_type(8)))  _Float16 v8h;
typedef __attribute__((ext_vector_type(8)))  float    v8f;

__device__ __forceinline__ v8f wmma_f16(v16h a, v16h b, v8f c) {
    // (neg_a, A, neg_b, B, c_mod, C, reuse_a, reuse_b)
    return __builtin_amdgcn_wmma_f32_16x16x32_f16(false, a, false, b, (short)0, c, false, false);
}

// A-operand fragment (16x32 f16 tile, row-major):
// lane r=lane&15 holds row r; half-wave selects col base 0 or 16; element e -> col base+e.
// Caller pre-adds (row*ld + colbase) into p.
__device__ __forceinline__ v16h loadA_h(const _Float16* p) {
    v8h a = *(const v8h*)p;
    v8h b = *(const v8h*)(p + 8);
    v16h r;
#pragma unroll
    for (int i = 0; i < 8; ++i) { r[i] = a[i]; r[i + 8] = b[i]; }
    return r;
}

__device__ __forceinline__ v16h loadA_f32cvt(const float* p) {
    v8f a = *(const v8f*)p;
    v8f b = *(const v8f*)(p + 8);
    v16h r;
#pragma unroll
    for (int i = 0; i < 8; ++i) { r[i] = (_Float16)a[i]; r[i + 8] = (_Float16)b[i]; }
    return r;
}

// B-operand fragment (32x16 f16, this code's contiguous K order, same as A: lanes 0-15 hold K=0..15, lanes 16-31
// K=16..31; i.e. 16 contiguous K per lane when reading B^T (NxK) row-major).
__device__ __forceinline__ v16h loadB_h(const _Float16* p) {
    return *(const v16h*)p;
}

// ---------------- patterns f32 -> f16 (plus transposed copy) ----------------
__global__ __launch_bounds__(256)
void cvt_patterns_kernel(const float* __restrict__ patterns,
                         _Float16* __restrict__ P16,
                         _Float16* __restrict__ PT16) {
    __shared__ float tile[32][33];
    const int f  = blockIdx.z;
    const int c0 = blockIdx.x * 32;
    const int n0 = blockIdx.y * 32;
    const int tx = threadIdx.x & 31;
    const int ty = threadIdx.x >> 5;   // 0..7
#pragma unroll
    for (int rr = 0; rr < 4; ++rr) {
        int row = rr * 8 + ty;         // class row within tile
        size_t idx = (((size_t)f * N_CLASSES) + (c0 + row)) * N_NEURONS + n0 + tx;
        float v = patterns[idx];
        tile[row][tx] = v;
        P16[idx] = (_Float16)v;
    }
    __syncthreads();
#pragma unroll
    for (int rr = 0; rr < 4; ++rr) {
        int row = rr * 8 + ty;         // neuron row within tile
        PT16[(((size_t)f * N_NEURONS) + (n0 + row)) * N_CLASSES + c0 + tx] =
            (_Float16)tile[tx][row];
    }
}

// ---------------- main Hopfield iteration kernel ----------------
// grid: (BATCH/16, N_FEATURES), block: 256 (8 waves of 32)
__global__ __launch_bounds__(256)
void hopfield_kernel(const float* __restrict__ x,
                     const _Float16* __restrict__ P16,
                     const _Float16* __restrict__ PT16,
                     float* __restrict__ out) {
    __shared__ _Float16 Sbuf[16 * N_CLASSES];   // softmax(S), 32 KB
    __shared__ _Float16 Tbuf[16 * N_NEURONS];   // T = S*P, 4 KB
    __shared__ float    maxbuf[8 * 16];
    __shared__ float    sumbuf[8 * 16];

    const int tid  = threadIdx.x;
    const int wave = tid >> 5;
    const int lane = tid & 31;
    const int r    = lane & 15;                 // row / N index within tile
    const int c0h  = (lane & 16) ? 16 : 0;      // A-fragment col base
    const int khi  = (lane & 16) ? 16 : 0;      // B-fragment K base
    const int rb   = (lane & 16) ? 8  : 0;      // C/D row base
    const int f    = blockIdx.y;
    const int b0   = blockIdx.x * 16;

    const _Float16* Pf  = P16  + (size_t)f * N_CLASSES * N_NEURONS;  // [class][n]
    const _Float16* PTf = PT16 + (size_t)f * N_NEURONS * N_CLASSES;  // [n][class]

    // ---- A fragments of X tile (16 batch rows x 128 neurons), f32 -> f16 ----
    v16h xa[4];
    {
        const float* xb = x + ((size_t)(b0 + r) * N_FEATURES + f) * N_NEURONS + c0h;
#pragma unroll
        for (int k = 0; k < 4; ++k) xa[k] = loadA_f32cvt(xb + k * 32);
    }

    v8f acc[8];
#pragma unroll
    for (int t = 0; t < 8; ++t)
#pragma unroll
        for (int i = 0; i < 8; ++i) acc[t][i] = 0.0f;

    // ---- init logits: L = X * P^T  (B^T view = P16, row-major, ld=128) ----
#pragma unroll
    for (int t = 0; t < 8; ++t) {
        const _Float16* pb = Pf + (size_t)(wave * 128 + t * 16 + r) * N_NEURONS + khi;
#pragma unroll
        for (int k = 0; k < 4; ++k)
            acc[t] = wmma_f16(xa[k], loadB_h(pb + k * 32), acc[t]);
    }

    for (int it = 0; it < N_ITER; ++it) {
        // ================= softmax over 1024 classes, rows = 16 batch =================
        float pm[8];
#pragma unroll
        for (int i = 0; i < 8; ++i) {
            float m = acc[0][i];
#pragma unroll
            for (int t = 1; t < 8; ++t) m = fmaxf(m, acc[t][i]);
            pm[i] = m;
        }
#pragma unroll
        for (int mask = 1; mask <= 8; mask <<= 1)
#pragma unroll
            for (int i = 0; i < 8; ++i) pm[i] = fmaxf(pm[i], __shfl_xor(pm[i], mask, 32));
        if (r == 0) {
#pragma unroll
            for (int i = 0; i < 8; ++i) maxbuf[wave * 16 + rb + i] = pm[i];
        }
        __syncthreads();
        float gm[8];
#pragma unroll
        for (int i = 0; i < 8; ++i) {
            float m = maxbuf[rb + i];
#pragma unroll
            for (int w = 1; w < 8; ++w) m = fmaxf(m, maxbuf[w * 16 + rb + i]);
            gm[i] = m;
        }
        float ps[8];
#pragma unroll
        for (int i = 0; i < 8; ++i) ps[i] = 0.0f;
#pragma unroll
        for (int t = 0; t < 8; ++t)
#pragma unroll
            for (int i = 0; i < 8; ++i) {
                float e = __expf(acc[t][i] - gm[i]);
                acc[t][i] = e;
                ps[i] += e;
            }
#pragma unroll
        for (int mask = 1; mask <= 8; mask <<= 1)
#pragma unroll
            for (int i = 0; i < 8; ++i) ps[i] += __shfl_xor(ps[i], mask, 32);
        if (r == 0) {
#pragma unroll
            for (int i = 0; i < 8; ++i) sumbuf[wave * 16 + rb + i] = ps[i];
        }
        __syncthreads();
        float inv[8];
#pragma unroll
        for (int i = 0; i < 8; ++i) {
            float s = sumbuf[rb + i];
#pragma unroll
            for (int w = 1; w < 8; ++w) s += sumbuf[w * 16 + rb + i];
            inv[i] = __frcp_rn(s);
        }
        // write normalized S (f16) into LDS, row-major [16][1024]
#pragma unroll
        for (int t = 0; t < 8; ++t) {
            int col = wave * 128 + t * 16 + r;
#pragma unroll
            for (int i = 0; i < 8; ++i)
                Sbuf[(rb + i) * N_CLASSES + col] = (_Float16)(acc[t][i] * inv[i]);
        }
        __syncthreads();

        // ================= T = S * P  (16x128, K=1024); wave owns cols n0..n0+15 ========
        v8f tacc;
#pragma unroll
        for (int i = 0; i < 8; ++i) tacc[i] = 0.0f;
        const _Float16* ptb = PTf + (size_t)(wave * 16 + r) * N_CLASSES + khi;
        const _Float16* sab = Sbuf + r * N_CLASSES + c0h;
#pragma unroll 8
        for (int kk = 0; kk < 32; ++kk) {
            v16h sa = loadA_h(sab + kk * 32);
            v16h pb = loadB_h(ptb + kk * 32);
            tacc = wmma_f16(sa, pb, tacc);
        }
        // store wave's T tile (f16) into LDS [16][128]
        {
            int col = wave * 16 + r;
#pragma unroll
            for (int i = 0; i < 8; ++i)
                Tbuf[(rb + i) * N_NEURONS + col] = (_Float16)tacc[i];
        }
        __syncthreads();

        // ================= L = T * P^T  (16x1024, K=128) =================
        v16h ta[4];
#pragma unroll
        for (int k = 0; k < 4; ++k) ta[k] = loadA_h(Tbuf + r * N_NEURONS + k * 32 + c0h);
#pragma unroll
        for (int t = 0; t < 8; ++t) {
#pragma unroll
            for (int i = 0; i < 8; ++i) acc[t][i] = 0.0f;
            const _Float16* pb = Pf + (size_t)(wave * 128 + t * 16 + r) * N_NEURONS + khi;
#pragma unroll
            for (int k = 0; k < 4; ++k)
                acc[t] = wmma_f16(ta[k], loadB_h(pb + k * 32), acc[t]);
        }
    }

    // ---- store logits tile: out[b][f][c], f32 ----
#pragma unroll
    for (int t = 0; t < 8; ++t) {
        int col = wave * 128 + t * 16 + r;
#pragma unroll
        for (int i = 0; i < 8; ++i) {
            size_t o = ((size_t)(b0 + rb + i) * N_FEATURES + f) * N_CLASSES + col;
            out[o] = acc[t][i];
        }
    }
}

extern "C" void kernel_launch(void* const* d_in, const int* in_sizes, int n_in,
                              void* d_out, int out_size, void* d_ws, size_t ws_size,
                              hipStream_t stream) {
    (void)in_sizes; (void)n_in; (void)out_size; (void)ws_size;
    const float* x        = (const float*)d_in[0];  // (256, 64, 128)
    const float* patterns = (const float*)d_in[1];  // (64, 1024, 128)
    float* out = (float*)d_out;                     // (256, 64, 1024)

    _Float16* P16  = (_Float16*)d_ws;                                      // [f][c][n]
    _Float16* PT16 = P16 + (size_t)N_FEATURES * N_CLASSES * N_NEURONS;     // [f][n][c]

    cvt_patterns_kernel<<<dim3(N_CLASSES / 32, N_NEURONS / 32, N_FEATURES),
                          256, 0, stream>>>(patterns, P16, PT16);
    hopfield_kernel<<<dim3(BATCH / 16, N_FEATURES), 256, 0, stream>>>(x, P16, PT16, out);
}
